// PAConv_54176717471777
// MI455X (gfx1250) — matrix operand order repaired and checked by reading, not compiled, not run on hardware
//
#include <hip/hip_runtime.h>
#include <hip/hip_bf16.h>
#include <stdint.h>

#define NPTS  4096
#define BATCH 4
#define NTOT  (NPTS * BATCH)
#define KNN   32

typedef __attribute__((ext_vector_type(16))) __bf16         v16bf;
typedef __attribute__((ext_vector_type(8)))  float          v8f;
typedef __attribute__((ext_vector_type(16))) unsigned short v16u16;
typedef __attribute__((ext_vector_type(8)))  unsigned short v8u16;

#define FLAG_BN      1
#define FLAG_RELU    2
#define FLAG_BBATCH  4   // bias indexed per batch: bias[b*O + m]
#define FLAG_OUTF32  8   // write f32 to (B,O,N) layout

#if defined(__has_builtin)
#if __has_builtin(__builtin_amdgcn_global_load_async_to_lds_b128) && \
    __has_builtin(__builtin_amdgcn_s_wait_asynccnt)
#define HAVE_ASYNC 1
#endif
#endif
#ifndef HAVE_ASYNC
#define HAVE_ASYNC 0
#endif

__device__ __forceinline__ unsigned short f2bf(float f) {
  unsigned u = __float_as_uint(f);
  u += 0x7FFFu + ((u >> 16) & 1u);            // round-to-nearest-even
  return (unsigned short)(u >> 16);
}
__device__ __forceinline__ float bf2f(unsigned short h) {
  return __uint_as_float(((unsigned)h) << 16);
}

// ---------------------------------------------------------------------------
// Weight pre-conversion: f32 slice -> bf16 row-major, zero-padded rows.
// ---------------------------------------------------------------------------
__global__ __launch_bounds__(256) void wcvt_kernel(
    const float* __restrict__ src, int lda, int col0,
    unsigned short* __restrict__ dst, int rows_src, int rows_dst, int cols) {
  const int t = blockIdx.x * 256 + threadIdx.x;
  if (t >= rows_dst * cols) return;
  const int rr = t / cols, cc = t - rr * cols;
  const float v = (rr < rows_src) ? src[(size_t)rr * lda + col0 + cc] : 0.0f;
  dst[t] = f2bf(v);
}

// ---------------------------------------------------------------------------
// kNN: top-32 smallest d2 per point, stable ties; batch points staged in LDS.
// ---------------------------------------------------------------------------
__global__ __launch_bounds__(256) void knn_kernel(const float* __restrict__ X,
                                                  int* __restrict__ idxb) {
  __shared__ float sx[NPTS], sy[NPTS], sz[NPTS];
  const int b = blockIdx.x >> 4;
  const int chunk = blockIdx.x & 15;
  const int tid = threadIdx.x;
  const float* xb = X + (size_t)b * 3 * NPTS;
  for (int i = tid; i < NPTS; i += 256) {
    sx[i] = xb[i];
    sy[i] = xb[NPTS + i];
    sz[i] = xb[2 * NPTS + i];
  }
  __syncthreads();
  const int n = chunk * 256 + tid;
  const float cx = sx[n], cy = sy[n], cz = sz[n];
  const float sqn = cx * cx + cy * cy + cz * cz;
  float bd[KNN]; int bi[KNN];
  #pragma unroll
  for (int k = 0; k < KNN; ++k) { bd[k] = 3.4e38f; bi[k] = 0; }
  for (int m = 0; m < NPTS; ++m) {
    const float px = sx[m], py = sy[m], pz = sz[m];
    const float d = sqn + (px * px + py * py + pz * pz)
                  - 2.0f * (cx * px + cy * py + cz * pz);
    if (d < bd[KNN - 1]) {
      int j = KNN - 1;
      while (j > 0 && d < bd[j - 1]) { bd[j] = bd[j - 1]; bi[j] = bi[j - 1]; --j; }
      bd[j] = d; bi[j] = m;
    }
  }
  int* op = idxb + ((size_t)b * NPTS + n) * KNN;
  #pragma unroll
  for (int k = 0; k < KNN; ++k) op[k] = bi[k];
}

// ---------------------------------------------------------------------------
// Edge conv: conv1(6->64)+bias, max over K neighbors, then BN. -> CAT[0:64]
// ---------------------------------------------------------------------------
__global__ __launch_bounds__(64) void edgeconv_kernel(
    const float* __restrict__ X, const int* __restrict__ idxb,
    const float* __restrict__ W, const float* __restrict__ bias,
    const float* __restrict__ gam, const float* __restrict__ bet,
    unsigned short* __restrict__ CAT) {
  const int p = blockIdx.x;
  const int b = p >> 12, n = p & (NPTS - 1);
  const int tid = threadIdx.x;
  __shared__ float nb[KNN][3];
  __shared__ float ct[3];
  const float* xb = X + (size_t)b * 3 * NPTS;
  if (tid < KNN) {
    const int j = idxb[(size_t)p * KNN + tid];
    nb[tid][0] = xb[j];
    nb[tid][1] = xb[NPTS + j];
    nb[tid][2] = xb[2 * NPTS + j];
  }
  if (tid < 3) ct[tid] = xb[tid * NPTS + n];
  __syncthreads();
  const int o = tid;
  const float w0 = W[o * 6 + 0], w1 = W[o * 6 + 1], w2 = W[o * 6 + 2];
  const float w3 = W[o * 6 + 3], w4 = W[o * 6 + 4], w5 = W[o * 6 + 5];
  const float c0 = ct[0], c1 = ct[1], c2 = ct[2];
  const float base = bias[o] + w3 * c0 + w4 * c1 + w5 * c2;
  float m = -3.4e38f;
  #pragma unroll 4
  for (int k = 0; k < KNN; ++k) {
    const float v = base + w0 * (nb[k][0] - c0) + w1 * (nb[k][1] - c1)
                  + w2 * (nb[k][2] - c2);
    m = fmaxf(m, v);
  }
  const float val = m * (gam[o] * rsqrtf(1.0f + 1e-5f)) + bet[o];
  CAT[(size_t)p * 1024 + o] = f2bf(val);
}

// ---------------------------------------------------------------------------
// Shared epilogue: acc[s][t] element v -> row m0+16s+8h+v, col n0+16t+r.
// ---------------------------------------------------------------------------
__device__ __forceinline__ void gemm_epilogue(
    v8f (&acc)[2][4], int m0, int n0, int r, int h,
    const float* __restrict__ bias,
    const float* __restrict__ gam, const float* __restrict__ bet,
    unsigned short* __restrict__ Dact, int ldd, int dcol,
    float* __restrict__ Dout, int O, int flags) {
  const float* bptr = bias;
  if (flags & FLAG_BBATCH) bptr += (size_t)(n0 >> 12) * O;
  const float bnmul = rsqrtf(1.0f + 1e-5f);
  #pragma unroll
  for (int s = 0; s < 2; ++s) {
    const int mb = m0 + (s << 4) + (h << 3);
    float bv[8], sc[8], sh[8];
    #pragma unroll
    for (int v = 0; v < 8; ++v) {
      const int mm = mb + v;
      const bool ok = mm < O;
      bv[v] = ok ? bptr[mm] : 0.0f;
      if (flags & FLAG_BN) {
        sc[v] = ok ? gam[mm] * bnmul : 0.0f;
        sh[v] = ok ? bet[mm] : 0.0f;
      }
    }
    #pragma unroll
    for (int t = 0; t < 4; ++t) {
      const int n = n0 + (t << 4) + r;
      float vals[8];
      #pragma unroll
      for (int v = 0; v < 8; ++v) {
        float x = acc[s][t][v] + bv[v];
        if (flags & FLAG_BN)   x = x * sc[v] + sh[v];
        if (flags & FLAG_RELU) x = fmaxf(x, 0.0f);
        vals[v] = x;
      }
      if (flags & FLAG_OUTF32) {
        const int bb = n >> 12, nn = n & (NPTS - 1);
        #pragma unroll
        for (int v = 0; v < 8; ++v) {
          const int mm = mb + v;
          if (mm < O) Dout[((size_t)bb * O + mm) * NPTS + nn] = vals[v];
        }
      } else {
        v8u16 pk;
        #pragma unroll
        for (int v = 0; v < 8; ++v) pk[v] = f2bf(vals[v]);
        *(v8u16*)(Dact + (size_t)n * ldd + dcol + mb) = pk;
      }
    }
  }
}

// ---------------------------------------------------------------------------
// Per-wave GEMM (small layers): 32x64 macro-tile, direct VMEM b128 loads.
// ---------------------------------------------------------------------------
__global__ __launch_bounds__(32) void gemm_wmma_kernel(
    const unsigned short* __restrict__ Wbf, int ldw,
    const float* __restrict__ bias,
    const float* __restrict__ gam, const float* __restrict__ bet,
    const unsigned short* __restrict__ Bact, int ldb, int bcol,
    unsigned short* __restrict__ Dact, int ldd, int dcol,
    float* __restrict__ Dout,
    int O, int C, int flags) {
  const int lane = threadIdx.x;
  const int r = lane & 15, h = lane >> 4;
  const int m0 = blockIdx.x << 5;
  const int n0 = blockIdx.y << 6;

  v8f acc[2][4];
  #pragma unroll
  for (int s = 0; s < 2; ++s)
    #pragma unroll
    for (int t = 0; t < 4; ++t)
      #pragma unroll
      for (int e = 0; e < 8; ++e) acc[s][t][e] = 0.0f;

  const unsigned short* ap0 = Wbf + (size_t)(m0 + r) * ldw + 8 * h;
  const unsigned short* ap1 = ap0 + (size_t)16 * ldw;
  const unsigned short* bp  = Bact + (size_t)(n0 + r) * ldb + bcol + 8 * h;
  const size_t bstride = (size_t)16 * ldb;

  for (int k0 = 0; k0 < C; k0 += 32) {
    const v8u16 a0lo = *(const v8u16*)(ap0 + k0);
    const v8u16 a0hi = *(const v8u16*)(ap0 + k0 + 16);
    const v8u16 a1lo = *(const v8u16*)(ap1 + k0);
    const v8u16 a1hi = *(const v8u16*)(ap1 + k0 + 16);
    const v16bf a0 = __builtin_bit_cast(v16bf,
        __builtin_shufflevector(a0lo, a0hi, 0,1,2,3,4,5,6,7,8,9,10,11,12,13,14,15));
    const v16bf a1 = __builtin_bit_cast(v16bf,
        __builtin_shufflevector(a1lo, a1hi, 0,1,2,3,4,5,6,7,8,9,10,11,12,13,14,15));
    #pragma unroll
    for (int t = 0; t < 4; ++t) {
      const v8u16 blo = *(const v8u16*)(bp + (size_t)t * bstride + k0);
      const v8u16 bhi = *(const v8u16*)(bp + (size_t)t * bstride + k0 + 16);
      const v16bf bm = __builtin_bit_cast(v16bf,
          __builtin_shufflevector(blo, bhi, 0,1,2,3,4,5,6,7,8,9,10,11,12,13,14,15));
      acc[0][t] = __builtin_amdgcn_wmma_f32_16x16x32_bf16(
          false, a0, false, bm, (short)0, acc[0][t], false, false);
      acc[1][t] = __builtin_amdgcn_wmma_f32_16x16x32_bf16(
          false, a1, false, bm, (short)0, acc[1][t], false, false);
    }
  }
  gemm_epilogue(acc, m0, n0, r, h, bias, gam, bet, Dact, ldd, dcol, Dout, O, flags);
}

// ---------------------------------------------------------------------------
// Block GEMM (large layers): 8 waves = 2(M)x4(N) -> 64x256 block tile.
// A(64x32) + B(256x32) staged in LDS per K-step, double-buffered; copies use
// global_load_async_to_lds_b128 (ASYNCcnt) when available. Row pitch 40
// shorts (80B) keeps ds_load_b128 fragment reads bank-spread & 16B-aligned.
// ---------------------------------------------------------------------------
#define PITCH 40              // shorts per LDS row (32 data + 8 pad)
#define AROWS 64
#define BROWS 256

#if HAVE_ASYNC
typedef int a_v4i __attribute__((vector_size(16)));
typedef __attribute__((address_space(1))) a_v4i* g_v4i_p;   // global 128b chunk
typedef __attribute__((address_space(3))) a_v4i* l_v4i_p;   // LDS 128b chunk
#define ASYNC_CP(gsrc, ldst)                                                   \
  __builtin_amdgcn_global_load_async_to_lds_b128((g_v4i_p)(gsrc),              \
                                                 (l_v4i_p)(ldst), 0, 0)
#define ASYNC_WAIT(n) __builtin_amdgcn_s_wait_asynccnt(n)
#else
#define ASYNC_CP(gsrc, ldst) (*(v8u16*)(ldst) = *(const v8u16*)(gsrc))
#define ASYNC_WAIT(n) ((void)0)
#endif

__global__ __launch_bounds__(256) void gemm_wmma_block_kernel(
    const unsigned short* __restrict__ Wbf, int ldw,
    const float* __restrict__ bias,
    const float* __restrict__ gam, const float* __restrict__ bet,
    const unsigned short* __restrict__ Bact, int ldb, int bcol,
    unsigned short* __restrict__ Dact, int ldd, int dcol,
    float* __restrict__ Dout,
    int O, int C, int flags) {
  __shared__ unsigned short smem[2][(AROWS + BROWS) * PITCH];   // 51.2 KB

  const int tid  = threadIdx.x;
  const int r    = tid & 15;
  const int h    = (tid >> 4) & 1;
  const int wave = tid >> 5;
  const int msub = wave & 1;        // 0..1
  const int nsub = wave >> 1;       // 0..3
  const int mblk = blockIdx.x << 6;                 // 64 rows per block
  const int nblk = blockIdx.y << 8;                 // 256 cols per block
  const int m0 = mblk + (msub << 5);
  const int n0 = nblk + (nsub << 6);

  // copy-slot mapping: slot -> (row, 8-element segment)
  const int a_row = tid >> 2, a_seg = tid & 3;
  const unsigned short* ag = Wbf + (size_t)(mblk + a_row) * ldw + a_seg * 8;
  const unsigned short* bg[4];
  int b_row[4], b_seg[4];
  #pragma unroll
  for (int q = 0; q < 4; ++q) {
    const int slot = q * 256 + tid;
    b_row[q] = slot >> 2; b_seg[q] = slot & 3;
    bg[q] = Bact + (size_t)(nblk + b_row[q]) * ldb + bcol + b_seg[q] * 8;
  }

  auto issue = [&](int k0, int buf) {
    unsigned short* s = &smem[buf][0];
    ASYNC_CP(ag + k0, s + a_row * PITCH + a_seg * 8);
    #pragma unroll
    for (int q = 0; q < 4; ++q)
      ASYNC_CP(bg[q] + k0, s + (AROWS + b_row[q]) * PITCH + b_seg[q] * 8);
  };

  v8f acc[2][4];
  #pragma unroll
  for (int s = 0; s < 2; ++s)
    #pragma unroll
    for (int t = 0; t < 4; ++t)
      #pragma unroll
      for (int e = 0; e < 8; ++e) acc[s][t][e] = 0.0f;

  const int nk = C >> 5;
  issue(0, 0);
  for (int i = 0; i < nk; ++i) {
    if (i + 1 < nk) {
      issue((i + 1) << 5, (i + 1) & 1);
      ASYNC_WAIT(5);          // the 5 just-issued remain; step-i copies done
    } else {
      ASYNC_WAIT(0);
    }
    __syncthreads();          // block-wide: buffer i ready

    const unsigned short* sm = &smem[i & 1][0];
    const unsigned short* ar0 = sm + (msub * 32 + r) * PITCH + 8 * h;
    const unsigned short* ar1 = ar0 + 16 * PITCH;
    const v8u16 a0lo = *(const v8u16*)(ar0);
    const v8u16 a0hi = *(const v8u16*)(ar0 + 16);
    const v8u16 a1lo = *(const v8u16*)(ar1);
    const v8u16 a1hi = *(const v8u16*)(ar1 + 16);
    const v16bf a0 = __builtin_bit_cast(v16bf,
        __builtin_shufflevector(a0lo, a0hi, 0,1,2,3,4,5,6,7,8,9,10,11,12,13,14,15));
    const v16bf a1 = __builtin_bit_cast(v16bf,
        __builtin_shufflevector(a1lo, a1hi, 0,1,2,3,4,5,6,7,8,9,10,11,12,13,14,15));
    #pragma unroll
    for (int t = 0; t < 4; ++t) {
      const unsigned short* br =
          sm + (AROWS + nsub * 64 + t * 16 + r) * PITCH + 8 * h;
      const v8u16 blo = *(const v8u16*)(br);
      const v8u16 bhi = *(const v8u16*)(br + 16);
      const v16bf bm = __builtin_bit_cast(v16bf,
          __builtin_shufflevector(blo, bhi, 0,1,2,3,4,5,6,7,8,9,10,11,12,13,14,15));
      acc[0][t] = __builtin_amdgcn_wmma_f32_16x16x32_bf16(
          false, a0, false, bm, (short)0, acc[0][t], false, false);
      acc[1][t] = __builtin_amdgcn_wmma_f32_16x16x32_bf16(
          false, a1, false, bm, (short)0, acc[1][t], false, false);
    }
    __syncthreads();          // done reading buffer i before it is rewritten
  }
  gemm_epilogue(acc, m0, n0, r, h, bias, gam, bet, Dact, ldd, dcol, Dout, O, flags);
}

// ---------------------------------------------------------------------------
// Global max-pool: g[b][o] = max_n G[(b*4096+n)][o] (G bf16, ld 1024)
// ---------------------------------------------------------------------------
__global__ __launch_bounds__(256) void gmax_kernel(
    const unsigned short* __restrict__ G, float* __restrict__ g) {
  const int o = blockIdx.x * 256 + threadIdx.x;
  const int b = blockIdx.y;
  const unsigned short* p = G + (size_t)b * NPTS * 1024 + o;
  float m = -3.4e38f;
  for (int n = 0; n < NPTS; ++n) m = fmaxf(m, bf2f(p[(size_t)n * 1024]));
  g[b * 1024 + o] = m;
}

// ---------------------------------------------------------------------------
// gv[b][o] = b7[o] + sum_c (W7[o,c]+W7[o,1024+c]) * g[b,c]
// ---------------------------------------------------------------------------
__global__ __launch_bounds__(256) void gv_kernel(
    const float* __restrict__ W7, const float* __restrict__ b7,
    const float* __restrict__ g, float* __restrict__ gv) {
  const int o = blockIdx.x * 256 + threadIdx.x;
  const int b = blockIdx.y;
  const float* wr = W7 + (size_t)o * 3008;
  const float* gb = g + b * 1024;
  float s = b7[o];
  for (int c = 0; c < 1024; ++c) s += (wr[c] + wr[1024 + c]) * gb[c];
  gv[b * 1024 + o] = s;
}

// ---------------------------------------------------------------------------
extern "C" void kernel_launch(void* const* d_in, const int* in_sizes, int n_in,
                              void* d_out, int out_size, void* d_ws, size_t ws_size,
                              hipStream_t stream) {
  (void)in_sizes; (void)n_in; (void)out_size; (void)ws_size;
  int i = 0;
  auto nx = [&]() { return (const float*)d_in[i++]; };

  const float* X = nx();                                        // (4,3,4096)
  const float *W1 = nx(), *b1 = nx(), *g1 = nx(), *bt1 = nx();  // conv1 + bn
  const float *W2 = nx(), *b2 = nx();
  const float *t2W[3], *t2b[3], *t2g[3], *t2bt[3];
  for (int k = 0; k < 3; ++k) { t2W[k]=nx(); t2b[k]=nx(); t2g[k]=nx(); t2bt[k]=nx(); }
  const float *W3 = nx(), *b3 = nx();
  const float *t3W[3], *t3b[3], *t3g[3], *t3bt[3];
  for (int k = 0; k < 3; ++k) { t3W[k]=nx(); t3b[k]=nx(); t3g[k]=nx(); t3bt[k]=nx(); }
  const float *W4 = nx(), *b4 = nx();
  const float *t4W[3], *t4b[3], *t4g[3], *t4bt[3];
  for (int k = 0; k < 3; ++k) { t4W[k]=nx(); t4b[k]=nx(); t4g[k]=nx(); t4bt[k]=nx(); }
  const float *W5 = nx(), *b5 = nx();
  const float *t5W[3], *t5b[3], *t5g[3], *t5bt[3];
  for (int k = 0; k < 3; ++k) { t5W[k]=nx(); t5b[k]=nx(); t5g[k]=nx(); t5bt[k]=nx(); }
  const float *W6 = nx(), *b6 = nx();
  const float *W7 = nx(), *b7 = nx();
  const float *h1W = nx(), *h1b = nx(), *h1g = nx(), *h1bt = nx();
  const float *h2W = nx(), *h2b = nx(), *h2g = nx(), *h2bt = nx();
  const float *h3W = nx(), *h3b = nx();

  char* ws = (char*)d_ws;
  size_t off = 0;
  auto carve = [&](size_t bytes) {
    void* p = ws + off;
    off = (off + bytes + 255) & ~(size_t)255;
    return p;
  };
  int*            idxb  = (int*)carve((size_t)NTOT * KNN * 4);
  float*          gbuf  = (float*)carve(4 * 1024 * 4);
  float*          gvbuf = (float*)carve(4 * 1024 * 4);
  unsigned short* CAT   = (unsigned short*)carve((size_t)NTOT * 1024 * 2);
  unsigned short* TA    = (unsigned short*)carve((size_t)NTOT * 512 * 2);
  unsigned short* TB    = (unsigned short*)carve((size_t)NTOT * 512 * 2);
  unsigned short* TC    = (unsigned short*)carve((size_t)NTOT * 1024 * 2);
  auto wbuf = [&](int rows, int cols) {
    return (unsigned short*)carve((size_t)rows * cols * 2);
  };
  unsigned short *wb2 = wbuf(64, 64), *wbt2[3], *wb3 = wbuf(128, 64), *wbt3[3];
  unsigned short *wb4 = wbuf(256, 128), *wbt4[3], *wb5 = wbuf(512, 256), *wbt5[3];
  for (int k = 0; k < 3; ++k) wbt2[k] = wbuf(64, 64);
  for (int k = 0; k < 3; ++k) wbt3[k] = wbuf(128, 128);
  for (int k = 0; k < 3; ++k) wbt4[k] = wbuf(256, 256);
  for (int k = 0; k < 3; ++k) wbt5[k] = wbuf(512, 512);
  unsigned short* wb6  = wbuf(1024, 1024);
  unsigned short* wb7  = wbuf(1024, 960);
  unsigned short* wbh1 = wbuf(512, 1024);
  unsigned short* wbh2 = wbuf(256, 512);
  unsigned short* wbh3 = wbuf(96, 256);     // padded 68 -> 96 rows

  auto wcvt = [&](const float* src, int lda, int col0, unsigned short* dst,
                  int rows_src, int rows_dst, int cols) {
    const int total = rows_dst * cols;
    wcvt_kernel<<<(total + 255) / 256, 256, 0, stream>>>(src, lda, col0, dst,
                                                         rows_src, rows_dst, cols);
  };
  // per-wave kernel (small layers)
  auto gemm = [&](const unsigned short* Wbf, int ldw, int Opad,
                  const float* bias, const float* gam, const float* bet,
                  const unsigned short* Bp, int ldb, int bcol,
                  unsigned short* Dp, int ldd, int dcol, float* Dout,
                  int O, int C, int flags) {
    dim3 grid((unsigned)(Opad / 32), (unsigned)(NTOT / 64));
    gemm_wmma_kernel<<<grid, 32, 0, stream>>>(Wbf, ldw, bias, gam, bet,
                                              Bp, ldb, bcol, Dp, ldd, dcol,
                                              Dout, O, C, flags);
  };
  // LDS-staged block kernel (large layers; Opad % 64 == 0)
  auto gemmB = [&](const unsigned short* Wbf, int ldw, int Opad,
                   const float* bias, const float* gam, const float* bet,
                   const unsigned short* Bp, int ldb, int bcol,
                   unsigned short* Dp, int ldd, int dcol, float* Dout,
                   int O, int C, int flags) {
    dim3 grid((unsigned)(Opad / 64), (unsigned)(NTOT / 256));
    gemm_wmma_block_kernel<<<grid, 256, 0, stream>>>(Wbf, ldw, bias, gam, bet,
                                                     Bp, ldb, bcol, Dp, ldd, dcol,
                                                     Dout, O, C, flags);
  };
  const int BR = FLAG_BN | FLAG_RELU;

  // weight pre-conversion
  wcvt(W2, 64, 0, wb2, 64, 64, 64);
  for (int k = 0; k < 3; ++k) wcvt(t2W[k], 64, 0, wbt2[k], 64, 64, 64);
  wcvt(W3, 64, 0, wb3, 128, 128, 64);
  for (int k = 0; k < 3; ++k) wcvt(t3W[k], 128, 0, wbt3[k], 128, 128, 128);
  wcvt(W4, 128, 0, wb4, 256, 256, 128);
  for (int k = 0; k < 3; ++k) wcvt(t4W[k], 256, 0, wbt4[k], 256, 256, 256);
  wcvt(W5, 256, 0, wb5, 512, 512, 256);
  for (int k = 0; k < 3; ++k) wcvt(t5W[k], 512, 0, wbt5[k], 512, 512, 512);
  wcvt(W6, 1024, 0, wb6, 1024, 1024, 1024);
  wcvt(W7, 3008, 2048, wb7, 1024, 1024, 960);   // x2..x5 slice only
  wcvt(h1W, 1024, 0, wbh1, 512, 512, 1024);
  wcvt(h2W, 512, 0, wbh2, 256, 256, 512);
  wcvt(h3W, 256, 0, wbh3, 68, 96, 256);

  knn_kernel<<<64, 256, 0, stream>>>(X, idxb);
  edgeconv_kernel<<<NTOT, 64, 0, stream>>>(X, idxb, W1, b1, g1, bt1, CAT); // x1

  // conv2 + t2 -> x2 at CAT[64:128]   (64-ch: per-wave kernel)
  gemm(wb2, 64, 64, b2, nullptr, nullptr, CAT, 1024, 0, TA, 64, 0, nullptr, 64, 64, 0);
  gemm(wbt2[0], 64, 64, t2b[0], t2g[0], t2bt[0], TA, 64, 0, TB, 64, 0, nullptr, 64, 64, BR);
  gemm(wbt2[1], 64, 64, t2b[1], t2g[1], t2bt[1], TB, 64, 0, TA, 64, 0, nullptr, 64, 64, BR);
  gemm(wbt2[2], 64, 64, t2b[2], t2g[2], t2bt[2], TA, 64, 0, CAT, 1024, 64, nullptr, 64, 64, BR);
  // conv3 + t3 -> x3 at CAT[128:256]  (128-ch: per-wave kernel)
  gemm(wb3, 64, 128, b3, nullptr, nullptr, CAT, 1024, 64, TA, 128, 0, nullptr, 128, 64, 0);
  gemm(wbt3[0], 128, 128, t3b[0], t3g[0], t3bt[0], TA, 128, 0, TB, 128, 0, nullptr, 128, 128, BR);
  gemm(wbt3[1], 128, 128, t3b[1], t3g[1], t3bt[1], TB, 128, 0, TA, 128, 0, nullptr, 128, 128, BR);
  gemm(wbt3[2], 128, 128, t3b[2], t3g[2], t3bt[2], TA, 128, 0, CAT, 1024, 128, nullptr, 128, 128, BR);
  // conv4 + t4 -> x4 at CAT[256:512]  (block kernel)
  gemmB(wb4, 128, 256, b4, nullptr, nullptr, CAT, 1024, 128, TA, 256, 0, nullptr, 256, 128, 0);
  gemmB(wbt4[0], 256, 256, t4b[0], t4g[0], t4bt[0], TA, 256, 0, TB, 256, 0, nullptr, 256, 256, BR);
  gemmB(wbt4[1], 256, 256, t4b[1], t4g[1], t4bt[1], TB, 256, 0, TA, 256, 0, nullptr, 256, 256, BR);
  gemmB(wbt4[2], 256, 256, t4b[2], t4g[2], t4bt[2], TA, 256, 0, CAT, 1024, 256, nullptr, 256, 256, BR);
  // conv5 + t5 -> x5 at CAT[512:1024] (block kernel)
  gemmB(wb5, 256, 512, b5, nullptr, nullptr, CAT, 1024, 256, TA, 512, 0, nullptr, 512, 256, 0);
  gemmB(wbt5[0], 512, 512, t5b[0], t5g[0], t5bt[0], TA, 512, 0, TB, 512, 0, nullptr, 512, 512, BR);
  gemmB(wbt5[1], 512, 512, t5b[1], t5g[1], t5bt[1], TB, 512, 0, TA, 512, 0, nullptr, 512, 512, BR);
  gemmB(wbt5[2], 512, 512, t5b[2], t5g[2], t5bt[2], TA, 512, 0, CAT, 1024, 512, nullptr, 512, 512, BR);

  // conv6 over cat -> TC, global max -> gbuf, broadcast term -> gvbuf
  gemmB(wb6, 1024, 1024, b6, nullptr, nullptr, CAT, 1024, 0, TC, 1024, 0, nullptr, 1024, 1024, 0);
  gmax_kernel<<<dim3(4, 4), 256, 0, stream>>>(TC, gbuf);
  gv_kernel<<<dim3(4, 4), 256, 0, stream>>>(W7, b7, gbuf, gvbuf);

  // conv7 = W7[:,2048:]·[x2..x5] + gv(b)  -> TC
  gemmB(wb7, 960, 1024, gvbuf, nullptr, nullptr, CAT, 1024, 64, TC, 1024, 0,
        nullptr, 1024, 960, FLAG_BBATCH);
  // heatmap head
  gemmB(wbh1, 1024, 512, h1b, h1g, h1bt, TC, 1024, 0, TA, 512, 0, nullptr, 512, 1024, BR);
  gemmB(wbh2, 512, 256, h2b, h2g, h2bt, TA, 512, 0, TB, 256, 0, nullptr, 256, 512, BR);
  gemm(wbh3, 256, 96, h3b, nullptr, nullptr, TB, 256, 0, nullptr, 0, 0,
       (float*)d_out, 68, 256, FLAG_OUTF32);
}
